// FM_13065290514484
// MI455X (gfx1250) — matrix operand order repaired and checked by reading, not compiled, not run on hardware
//
#include <hip/hip_runtime.h>

// ---- CDNA5 WMMA vector types -------------------------------------------------
typedef __attribute__((ext_vector_type(16))) __bf16       v16bf;
typedef __attribute__((ext_vector_type(8)))  float        v8f;

// ---- Problem geometry (B=262144, F=512, K=64) --------------------------------
#define FM_F            512
#define FM_K            64
#define NSTEP           16                 // F / 32 (K-dim per WMMA = 32)
#define NTILE           4                  // K / 16 (N tiles)
#define FM_WAVES        8
#define ROWS_PER_WG     (FM_WAVES * 16)    // 128
#define PACK_U32        (NSTEP * NTILE * 32 * 8)      // 16384 u32 per packed matrix (64KB)
#define LDS_W_OFF       (2 * PACK_U32)                // 32768: w[512] floats
#define LDS_SCR_OFF     (LDS_W_OFF + FM_F)            // 33280: per-wave scratch
#define LDS_TOTAL_U32   (LDS_SCR_OFF + 2 * FM_WAVES * 16)   // 33536
#define LDS_BYTES       (LDS_TOTAL_U32 * 4)           // 134144 B <= 320KB/WGP

// ---- helpers -----------------------------------------------------------------
__device__ __forceinline__ unsigned pack_bf16x2(float lo, float hi) {
    union { __bf16 h[2]; unsigned u; } p;
    p.h[0] = (__bf16)lo;
    p.h[1] = (__bf16)hi;
    return p.u;
}

union BFrag {
    uint4 q[2];
    v16bf v;
};

// =============================================================================
// Prep kernel: convert bw (f32 [512,64], row-major, k fastest) into bf16 bw and
// bf16 bw^2, laid out in the contiguous-k (k = 16h + s) B-matrix (32x16, bf16) order used here:
//   entry e = ((s*4 + t)*32 + L)*8 + v
//   lane L: column n = t*16 + (L&15); kbase = (L<16) ? 0 : 16
//   VGPR v holds K pair (kbase+2v, kbase+2v+1) at f = s*32 + kk
// =============================================================================
__global__ void fm_prep_kernel(const float* __restrict__ bw,
                               unsigned* __restrict__ packA,   // bf16(bw)
                               unsigned* __restrict__ packQ) { // bf16(bw^2)
    int e = blockIdx.x * 256 + threadIdx.x;
    if (e >= PACK_U32) return;
    int v = e & 7;
    int L = (e >> 3) & 31;
    int t = (e >> 8) & 3;
    int s = e >> 10;
    int kb = (L & 16) ? 16 : 0;
    int kk = kb + 2 * v;
    int f  = s * 32 + kk;
    int n  = t * 16 + (L & 15);
    float v0 = bw[(size_t)f * FM_K + n];
    float v1 = bw[(size_t)(f + 1) * FM_K + n];
    packA[e] = pack_bf16x2(v0, v1);
    packQ[e] = pack_bf16x2(v0 * v0, v1 * v1);
}

// ---- pipelined fetch helpers -------------------------------------------------
__device__ __forceinline__ void load_frags(const uint4* bbase, int s,
                                           BFrag* bf, BFrag* qf) {
    #pragma unroll
    for (int t = 0; t < NTILE; ++t) {
        const uint4* bp = bbase + (s * NTILE + t) * 64;   // uint4 units
        const uint4* qp = bp + PACK_U32 / 4;
        bf[t].q[0] = bp[0];  bf[t].q[1] = bp[1];          // 2x ds_load_b128
        qf[t].q[0] = qp[0];  qf[t].q[1] = qp[1];
    }
}

__device__ __forceinline__ void load_x(const float* xrow, int s, int bk, float4* cx) {
    const float4* p0 = (const float4*)(xrow + s * 32 + bk);
    cx[0] = p0[0];  cx[1] = p0[1];
    const float4* p1 = (const float4*)(xrow + s * 32 + bk + 8);
    cx[2] = p1[0];  cx[3] = p1[1];
}

// =============================================================================
// Main kernel: 8 waves * 16 rows = 128 rows / workgroup.
// __launch_bounds__(256, 2): 2 waves/SIMD resident -> up to 512 VGPRs/wave.
// Fully unrolled, double-buffered F-loop: fragments for step s+1 are in flight
// while the 8-WMMA chain of step s executes.
// =============================================================================
__global__ void __launch_bounds__(256, 2)
fm_main_kernel(const float* __restrict__ x,
               const float* __restrict__ w0p,
               const float* __restrict__ w,
               const unsigned* __restrict__ packed, // [2*PACK_U32]
               float* __restrict__ out) {
    extern __shared__ unsigned int lds[];
    const int tid = threadIdx.x;

    // ---- stage both packed B matrices (128KB) + w (2KB) into LDS ------------
    {
        const uint4* src = (const uint4*)packed;
        uint4*       dst = (uint4*)lds;
        #pragma unroll
        for (int i = 0; i < (2 * PACK_U32) / 4 / 256; ++i)   // 32 iters
            dst[tid + i * 256] = src[tid + i * 256];
        if (tid < FM_F / 4)
            ((uint4*)(lds + LDS_W_OFF))[tid] = ((const uint4*)w)[tid];
    }
    __syncthreads();

    const int  lane = tid & 31;
    const int  wv   = tid >> 5;
    const int  half = lane >> 4;       // 0: lanes 0-15, 1: lanes 16-31
    const int  mrow = lane & 15;
    const int  bk   = half * 16;       // A-layout K sub-offset
    const long rowBase = (long)blockIdx.x * ROWS_PER_WG + wv * 16;

    const float* xrow = x + (rowBase + mrow) * FM_F;
    const float* wlds = (const float*)(lds + LDS_W_OFF);
    const uint4* bbase = (const uint4*)lds + lane * 2;   // lane*32B

    v8f acc[NTILE], qcc[NTILE];
    v8f vzero = {};
    #pragma unroll
    for (int t = 0; t < NTILE; ++t) { acc[t] = vzero; qcc[t] = vzero; }
    float lin = 0.0f;

    // ---- software pipeline: prologue fetch of step 0 ------------------------
    BFrag  bf[2][NTILE], qf[2][NTILE];
    float4 cx[2][4];
    load_frags(bbase, 0, bf[0], qf[0]);
    load_x(xrow, 0, bk, cx[0]);

    #pragma unroll
    for (int s = 0; s < NSTEP; ++s) {
        const int cur = s & 1, nxt = cur ^ 1;

        // fetch step s+1 while step s computes (stays in flight across WMMAs)
        if (s + 1 < NSTEP) {
            load_frags(bbase, s + 1, bf[nxt], qf[nxt]);
            load_x(xrow, s + 1, bk, cx[nxt]);
        }
        __builtin_prefetch(xrow + s * 32 + 256, 0, 3);   // 8 steps ahead

        const float4 a0 = cx[cur][0], a1 = cx[cur][1];
        const float4 b0 = cx[cur][2], b1 = cx[cur][3];
        float c[16] = {a0.x, a0.y, a0.z, a0.w, a1.x, a1.y, a1.z, a1.w,
                       b0.x, b0.y, b0.z, b0.w, b1.x, b1.y, b1.z, b1.w};

        v16bf A, A2;
        const float* wp = wlds + s * 32 + bk;
        #pragma unroll
        for (int i = 0; i < 8; ++i) {
            A[i]     = (__bf16)c[i];
            A[i + 8] = (__bf16)c[i + 8];
            float q0 = c[i] * c[i];
            float q1 = c[i + 8] * c[i + 8];
            A2[i]     = (__bf16)q0;
            A2[i + 8] = (__bf16)q1;
            lin += c[i] * wp[i] + c[i + 8] * wp[8 + i];
        }

        #pragma unroll
        for (int t = 0; t < NTILE; ++t) {
            acc[t] = __builtin_amdgcn_wmma_f32_16x16x32_bf16(
                         false, A,  false, bf[cur][t].v, (short)0, acc[t], false, false);
            qcc[t] = __builtin_amdgcn_wmma_f32_16x16x32_bf16(
                         false, A2, false, qf[cur][t].v, (short)0, qcc[t], false, false);
        }
    }

    // ---- epilogue: inter_row = sum_k (xv_k^2 - x2v2_k) ----------------------
    // C layout: VGPR r -> row (half*8 + r), column n = mrow + 16*t (t folded in-lane)
    float s8[8];
    #pragma unroll
    for (int r = 0; r < 8; ++r) {
        float v = 0.0f;
        #pragma unroll
        for (int t = 0; t < NTILE; ++t)
            v += acc[t][r] * acc[t][r] - qcc[t][r];
        s8[r] = v;
    }
    // reduce across the 16 lanes of each half (columns)
    #pragma unroll
    for (int m = 1; m <= 8; m <<= 1) {
        #pragma unroll
        for (int r = 0; r < 8; ++r) s8[r] += __shfl_xor(s8[r], m, 32);
    }
    // linear term: lanes L and L+16 hold complementary halves of the same row
    lin += __shfl_xor(lin, 16, 32);

    float* interL = (float*)(lds + LDS_SCR_OFF) + wv * 16;
    float* linL   = (float*)(lds + LDS_SCR_OFF) + FM_WAVES * 16 + wv * 16;
    if (lane == 0) {
        #pragma unroll
        for (int r = 0; r < 8; ++r) interL[r] = s8[r];
    }
    if (lane == 16) {
        #pragma unroll
        for (int r = 0; r < 8; ++r) interL[8 + r] = s8[r];
    }
    if (lane < 16) linL[mrow] = lin;
    __syncthreads();

    if (lane < 16) {
        float a = linL[mrow] + w0p[0] + 0.5f * interL[mrow];
        out[rowBase + mrow] = 1.0f / (1.0f + __expf(-a));
    }
}

// =============================================================================
extern "C" void kernel_launch(void* const* d_in, const int* in_sizes, int n_in,
                              void* d_out, int out_size, void* d_ws, size_t ws_size,
                              hipStream_t stream) {
    const float* x  = (const float*)d_in[0];   // [B, 512]
    const float* w0 = (const float*)d_in[1];   // [1]
    const float* w  = (const float*)d_in[2];   // [512]
    const float* bw = (const float*)d_in[3];   // [512, 64]
    float*       out = (float*)d_out;

    unsigned* packA = (unsigned*)d_ws;             // 64KB
    unsigned* packQ = packA + PACK_U32;            // 64KB

    const int Brows = in_sizes[0] / FM_F;          // 262144

    fm_prep_kernel<<<(PACK_U32 + 255) / 256, 256, 0, stream>>>(bw, packA, packQ);

    const int grid = Brows / ROWS_PER_WG;          // 2048
    fm_main_kernel<<<grid, 256, LDS_BYTES, stream>>>(x, w0, w, packA, out);
}
